// VAE_29119878267085
// MI455X (gfx1250) — matrix operand order repaired and checked by reading, not compiled, not run on hardware
//
#include <hip/hip_runtime.h>
#include <hip/hip_bf16.h>

// ---------------------------------------------------------------------------
// VQ-VAE forward for MI455X (gfx1250, wave32, WMMA).
// Convs run as implicit GEMMs on v_wmma_f32_16x16x32_bf16 with LDS-staged,
// branch-free im2col patches shared by all waves of a workgroup; each wave
// computes NT independent 16x16 N-tiles per A-fragment (amortizes DS reads
// and interleaves independent WMMA chains into the hazard slots).
// ---------------------------------------------------------------------------

typedef __bf16 bf16_t;
typedef __attribute__((ext_vector_type(16))) __bf16 v16bf;
typedef __attribute__((ext_vector_type(2)))  __bf16 v2bf;
typedef __attribute__((ext_vector_type(8)))  float  v8f;
typedef __attribute__((ext_vector_type(8)))  unsigned int v8u;

static __device__ __forceinline__ float silu_f(float v) {
  return v / (1.f + __expf(-v));
}
static __device__ __forceinline__ int clampi(int v, int lo, int hi) {
  return v < lo ? lo : (v > hi ? hi : v);
}

// ---------------------------------------------------------------------------
// Strided conv (k=4, s=2, p=1) as implicit GEMM.
//   M = 16 consecutive ox (fixed b,oy); N = 16*NW*NT channels per block;
//   K = CIN*16 (k = cin*16 + ky*4 + kx == flattened OIHW).
// LDS patch: [ROWS=KPAD/4][PW=36] bf16, row = cin*4+ky, col xx = ix-(2*ox0-1).
// Branch-free fill: clamped addresses + value select (no exec-mask branches).
// A-frag K-pair (k,k+1) == u32 at pat32[row*18 + r + jj] (always 4B aligned).
// ---------------------------------------------------------------------------
template<int CIN, int COUT, int HIN, int WIN, int NW, int NT>
__global__ __launch_bounds__(32 * NW)
void conv_s2_wmma(const float* __restrict__ in, const bf16_t* __restrict__ wbf,
                  const float* __restrict__ bias, float* __restrict__ out)
{
  constexpr int HOUT = HIN / 2, WOUT = WIN / 2;
  constexpr int KTOT = CIN * 16;
  constexpr int KPAD = (KTOT + 31) & ~31;
  constexpr int KSTEPS = KPAD / 32;
  constexpr int ROWS = KPAD / 4;          // cin*4+ky rows (incl. K-pad rows)
  constexpr int PW = 36;                  // padded patch width (even)
  __shared__ bf16_t patch[ROWS * PW];

  const int tid  = threadIdx.x;
  const int lane = tid & 31;
  const int wv   = tid >> 5;
  const int ox0  = blockIdx.x * 16;
  const int b    = blockIdx.y / HOUT;
  const int oy   = blockIdx.y % HOUT;
  const int n0   = blockIdx.z * (16 * NW * NT) + wv * (16 * NT);

  const int iy0 = 2 * oy - 1;             // pad = 1
  const int ixb = 2 * ox0 - 1;

  // Cooperative branch-free fill (zero-pads borders and K-padding rows).
  for (int idx = tid; idx < ROWS * PW; idx += 32 * NW) {
    const int xx  = idx % PW;
    const int row = idx / PW;
    const int cin = row >> 2;
    const int iy  = iy0 + (row & 3);
    const int ix  = ixb + xx;
    const int cc  = cin < CIN ? cin : CIN - 1;
    const float v = in[((size_t)(b * CIN + cc) * HIN + clampi(iy, 0, HIN - 1)) * WIN
                       + clampi(ix, 0, WIN - 1)];
    const bool ok = (cin < CIN) & (iy >= 0) & (iy < HIN) & (ix >= 0) & (ix < WIN);
    patch[idx] = (bf16_t)(ok ? v : 0.f);
  }
  __syncthreads();

  const unsigned int* pat32 = reinterpret_cast<const unsigned int*>(patch);
  const int r    = lane & 15;
  const int koff = (lane < 16) ? 0 : 16;
  const bf16_t* wrow0 = wbf + (size_t)(n0 + r) * KPAD + ((lane < 16) ? 0 : 16);
  const bf16_t* wrow1 = wrow0 + (size_t)16 * KPAD;

  v8f acc0 = {}, acc1 = {};
  for (int ks = 0; ks < KSTEPS; ++ks) {
    const int kbase = ks * 32;
    v16bf bf0 = *reinterpret_cast<const v16bf*>(wrow0 + kbase);
    if (ks + 1 < KSTEPS) __builtin_prefetch(wrow0 + kbase + 32, 0, 1);

    v8u a;
#pragma unroll
    for (int half = 0; half < 2; ++half) {
      const int k0   = kbase + koff + half * 8;
      const int row0 = k0 >> 2;           // cin*4 + ky0 (k0 multiple of 4)
#pragma unroll
      for (int g = 0; g < 2; ++g)
#pragma unroll
        for (int jj = 0; jj < 2; ++jj)
          a[half * 4 + g * 2 + jj] = pat32[(row0 + g) * (PW / 2) + r + jj];
    }
    const v16bf av = __builtin_bit_cast(v16bf, a);
    acc0 = __builtin_amdgcn_wmma_f32_16x16x32_bf16(false, av, false, bf0,
                                                   (short)0, acc0, false, false);
    if constexpr (NT == 2) {
      v16bf bf1 = *reinterpret_cast<const v16bf*>(wrow1 + kbase);
      acc1 = __builtin_amdgcn_wmma_f32_16x16x32_bf16(false, av, false, bf1,
                                                     (short)0, acc1, false, false);
    }
  }

  // C/D layout: lane holds N = lane&15, M = vgpr + (lane<16 ? 0 : 8).
  const int moff = (lane < 16) ? 0 : 8;
  {
    const int   n  = n0 + r;
    const float bv = bias[n];
    float* op = out + ((size_t)(b * COUT + n) * HOUT + oy) * WOUT + ox0 + moff;
#pragma unroll
    for (int i = 0; i < 8; ++i) op[i] = acc0[i] + bv;
  }
  if constexpr (NT == 2) {
    const int   n  = n0 + 16 + r;
    const float bv = bias[n];
    float* op = out + ((size_t)(b * COUT + n) * HOUT + oy) * WOUT + ox0 + moff;
#pragma unroll
    for (int i = 0; i < 8; ++i) op[i] = acc1[i] + bv;
  }
}

// ---------------------------------------------------------------------------
// ConvTranspose2d (k=4, s=2, p=1) via 4 output-parity quadrants; each quadrant
// is a stride-1 implicit GEMM with K = 4*CIN, taps ordered by ascending (iy,ix):
//   k = cin*4 + ty*2 + tx,  iy = m + dys[0] + ty,  ix = j + dxs[0] + tx,
//   dys = py ? {0,+1} : {-1,0},  dxs likewise with px.
// LDS patch pre-packs bf16 x-pairs: tpatch[(cin*2+ty)*16 + r] = {x=r, x=r+1}.
// A-frag pair for (k,k+1): u32 at tpatch[(k>>1)*16 + r]  (k even => k>>1 = cin*2+ty).
// ---------------------------------------------------------------------------
template<int CIN, int COUT, int COUTP, int HIN, int WIN, int ACT, int NW, int NT>
__global__ __launch_bounds__(32 * NW)
void convt_s2_wmma(const float* __restrict__ in, const bf16_t* __restrict__ wq4,
                   const float* __restrict__ bias, float* __restrict__ out)
{
  constexpr int HOUT = HIN * 2, WOUT = WIN * 2;
  constexpr int KTOT = CIN * 4;           // multiple of 32 for all layers
  constexpr int KSTEPS = KTOT / 32;
  constexpr int NTB = 16 * NW * NT;       // N channels covered per block
  constexpr int ZT = COUTP / NTB;         // z-tiles per quadrant
  __shared__ unsigned int tpatch[CIN * 2 * 16];

  const int tid  = threadIdx.x;
  const int lane = tid & 31;
  const int wv   = tid >> 5;
  const int j0   = blockIdx.x * 16;
  const int b    = blockIdx.y / HIN;
  const int m    = blockIdx.y % HIN;      // input-grid row
  const int q    = blockIdx.z / ZT;       // quadrant
  const int n0   = (blockIdx.z % ZT) * NTB + wv * (16 * NT);
  const int py = q >> 1, px = q & 1;
  const int dy0 = py ? 0 : -1;            // dys[0]
  const int dx0 = px ? 0 : -1;            // dxs[0]

  // Cooperative branch-free fill of packed x-pairs.
  for (int idx = tid; idx < CIN * 2 * 16; idx += 32 * NW) {
    const int rr  = idx & 15;
    const int row = idx >> 4;             // cin*2 + ty
    const int cin = row >> 1;
    const int iy  = m + dy0 + (row & 1);
    const int ixp = j0 + dx0 + rr;
    const bool yok = (iy >= 0) & (iy < HIN);
    const float* rp = in + ((size_t)(b * CIN + cin) * HIN + clampi(iy, 0, HIN - 1)) * WIN;
    float v0 = rp[clampi(ixp, 0, WIN - 1)];
    float v1 = rp[clampi(ixp + 1, 0, WIN - 1)];
    v0 = (yok && ixp >= 0 && ixp < WIN) ? v0 : 0.f;
    v1 = (yok && (ixp + 1) >= 0 && (ixp + 1) < WIN) ? v1 : 0.f;
    v2bf p; p[0] = (bf16_t)v0; p[1] = (bf16_t)v1;
    tpatch[idx] = __builtin_bit_cast(unsigned int, p);
  }
  __syncthreads();

  const int r    = lane & 15;
  const int koff = (lane < 16) ? 0 : 16;
  const bf16_t* wrow0 = wq4 + (size_t)q * COUTP * KTOT
                      + (size_t)(n0 + r) * KTOT + ((lane < 16) ? 0 : 16);
  const bf16_t* wrow1 = wrow0 + (size_t)16 * KTOT;

  v8f acc0 = {}, acc1 = {};
  for (int ks = 0; ks < KSTEPS; ++ks) {
    const int kbase = ks * 32;
    v16bf bf0 = *reinterpret_cast<const v16bf*>(wrow0 + kbase);
    if (ks + 1 < KSTEPS) __builtin_prefetch(wrow0 + kbase + 32, 0, 1);

    v8u a;
#pragma unroll
    for (int half = 0; half < 2; ++half) {
      const int k0 = kbase + koff + half * 8;
#pragma unroll
      for (int p = 0; p < 4; ++p) {
        const int k = k0 + 2 * p;         // even: k>>1 == cin*2 + ty
        a[half * 4 + p] = tpatch[(k >> 1) * 16 + r];
      }
    }
    const v16bf av = __builtin_bit_cast(v16bf, a);
    acc0 = __builtin_amdgcn_wmma_f32_16x16x32_bf16(false, av, false, bf0,
                                                   (short)0, acc0, false, false);
    if constexpr (NT == 2) {
      v16bf bf1 = *reinterpret_cast<const v16bf*>(wrow1 + kbase);
      acc1 = __builtin_amdgcn_wmma_f32_16x16x32_bf16(false, av, false, bf1,
                                                     (short)0, acc1, false, false);
    }
  }

  const int moff = (lane < 16) ? 0 : 8;
  const int oy   = 2 * m + py;
#pragma unroll
  for (int t = 0; t < NT; ++t) {
    const int n = n0 + t * 16 + r;
    if (n < COUT) {
      const float bv = bias[n];
      float* op = out + ((size_t)(b * COUT + n) * HOUT + oy) * WOUT + px;
#pragma unroll
      for (int i = 0; i < 8; ++i) {
        float v = (t == 0 ? acc0[i] : acc1[i]) + bv;
        v = (ACT == 0) ? silu_f(v) : tanhf(v);
        op[(size_t)2 * (j0 + moff + i)] = v;  // ox = 2*j + px
      }
    }
  }
}

// ---------------------------------------------------------------------------
// BatchNorm training-mode stats (biased variance) + fused normalize/SiLU.
// ---------------------------------------------------------------------------
template<int C, int HW>
__global__ void bn_stats(const float* __restrict__ x, float* __restrict__ sums)
{
  const int c = blockIdx.x, b = blockIdx.y;
  const float* p = x + ((size_t)b * C + c) * HW;
  float s = 0.f, s2 = 0.f;
  for (int i = threadIdx.x; i < HW; i += 256) { float v = p[i]; s += v; s2 += v * v; }
  __shared__ float sh0[256], sh1[256];
  sh0[threadIdx.x] = s; sh1[threadIdx.x] = s2;
  __syncthreads();
  for (int st = 128; st > 0; st >>= 1) {
    if (threadIdx.x < st) {
      sh0[threadIdx.x] += sh0[threadIdx.x + st];
      sh1[threadIdx.x] += sh1[threadIdx.x + st];
    }
    __syncthreads();
  }
  if (threadIdx.x == 0) {
    atomicAdd(&sums[c], sh0[0]);
    atomicAdd(&sums[C + c], sh1[0]);
  }
}

template<int C, int HW, int BATCH>
__global__ void bn_apply_silu(float* __restrict__ x, const float* __restrict__ sums,
                              const float* __restrict__ g, const float* __restrict__ bb)
{
  const size_t total = (size_t)BATCH * C * HW;
  const size_t i = (size_t)blockIdx.x * blockDim.x + threadIdx.x;
  if (i >= total) return;
  const int c = (int)((i / HW) % C);
  constexpr float inv = 1.f / ((float)BATCH * (float)HW);
  const float mean = sums[c] * inv;
  const float var  = sums[C + c] * inv - mean * mean;
  const float v = (x[i] - mean) * rsqrtf(var + 1e-5f) * g[c] + bb[c];
  x[i] = silu_f(v);
}

// ---------------------------------------------------------------------------
// Vector quantizer: one wave per latent vector (D=64 -> 2 channels/lane).
// Forward value of straight-through is z_q; vq_loss = 1.25*mean((z_q-z_e)^2).
// ---------------------------------------------------------------------------
__global__ __launch_bounds__(32)
void vq_argmin(const float* __restrict__ ze, const float* __restrict__ cb,
               float* __restrict__ zq, float* __restrict__ sse)
{
  const int lane = threadIdx.x;
  const int vec  = blockIdx.x;                 // 0 .. 32*32*32-1
  const int b    = vec >> 10;                  // H*W = 1024
  const int pix  = vec & 1023;
  const size_t base = ((size_t)b * 64) * 1024 + pix;
  const float z0 = ze[base + (size_t)lane * 1024];
  const float z1 = ze[base + (size_t)(lane + 32) * 1024];

  float best = 3.4e38f; int bestc = 0;
  for (int c = 0; c < 512; ++c) {
    const float a0 = cb[c * 64 + lane];
    const float a1 = cb[c * 64 + lane + 32];
    float d = (z0 - a0) * (z0 - a0) + (z1 - a1) * (z1 - a1);
#pragma unroll
    for (int off = 16; off > 0; off >>= 1) d += __shfl_xor(d, off, 32);
    if (d < best) { best = d; bestc = c; }     // strict < keeps first min (argmin)
  }
  const float q0 = cb[bestc * 64 + lane];
  const float q1 = cb[bestc * 64 + lane + 32];
  zq[base + (size_t)lane * 1024]        = q0;
  zq[base + (size_t)(lane + 32) * 1024] = q1;

  float e = (q0 - z0) * (q0 - z0) + (q1 - z1) * (q1 - z1);
#pragma unroll
  for (int off = 16; off > 0; off >>= 1) e += __shfl_xor(e, off, 32);
  if (lane == 0) atomicAdd(sse, e);
}

// ---------------------------------------------------------------------------
// Weight preparation (f32 -> bf16 GEMM layouts), re-run every launch.
// ---------------------------------------------------------------------------
template<int CIN, int COUT>
__global__ void prep_conv_w(const float* __restrict__ w, bf16_t* __restrict__ o)
{
  constexpr int KTOT = CIN * 16;
  constexpr int KPAD = (KTOT + 31) & ~31;
  const int i = blockIdx.x * 256 + threadIdx.x;
  if (i >= COUT * KPAD) return;
  const int n = i / KPAD, k = i % KPAD;
  o[i] = (bf16_t)((k < KTOT) ? w[(size_t)n * KTOT + k] : 0.f);
}

template<int CIN, int COUT, int COUTP>
__global__ void prep_convt_w(const float* __restrict__ w, bf16_t* __restrict__ o)
{
  constexpr int K = CIN * 4;
  const int i = blockIdx.x * 256 + threadIdx.x;
  if (i >= 4 * COUTP * K) return;
  const int q   = i / (COUTP * K);
  const int rem = i % (COUTP * K);
  const int n = rem / K, k = rem % K;
  const int py = q >> 1, px = q & 1;
  const int cin = k >> 2, ty = (k >> 1) & 1, tx = k & 1;
  // Taps ordered by ascending (iy, ix):  ky = py+1-2*dy,  dy = dys[ty].
  const int ky = py ? (ty ? 0 : 2) : (ty ? 1 : 3);
  const int kx = px ? (tx ? 0 : 2) : (tx ? 1 : 3);
  float v = 0.f;
  if (n < COUT) v = w[(((size_t)cin * COUT + n) * 4 + ky) * 4 + kx];  // torch [in][out][kh][kw]
  o[i] = (bf16_t)v;
}

__global__ void zero_ws(float* __restrict__ p, int n)
{
  const int i = blockIdx.x * 256 + threadIdx.x;
  if (i < n) p[i] = 0.f;
}

__global__ void finish_loss(const float* __restrict__ sse, float* __restrict__ out)
{
  if (threadIdx.x == 0) out[0] = 1.25f * sse[0] * (1.f / 2097152.f);
}

// ---------------------------------------------------------------------------
extern "C" void kernel_launch(void* const* d_in, const int* in_sizes, int n_in,
                              void* d_out, int out_size, void* d_ws, size_t ws_size,
                              hipStream_t stream)
{
  (void)in_sizes; (void)n_in; (void)out_size; (void)ws_size;
  const float* x   = (const float*)d_in[0];
  const float* ew1 = (const float*)d_in[1];
  const float* eb1 = (const float*)d_in[2];
  const float* bg1 = (const float*)d_in[3];
  const float* bb1 = (const float*)d_in[4];
  const float* ew2 = (const float*)d_in[5];
  const float* eb2 = (const float*)d_in[6];
  const float* bg2 = (const float*)d_in[7];
  const float* bb2 = (const float*)d_in[8];
  const float* ew3 = (const float*)d_in[9];
  const float* eb3 = (const float*)d_in[10];
  const float* cb  = (const float*)d_in[11];
  const float* dw1 = (const float*)d_in[12];
  const float* db1 = (const float*)d_in[13];
  const float* dw2 = (const float*)d_in[14];
  const float* db2 = (const float*)d_in[15];
  const float* dw3 = (const float*)d_in[16];
  const float* db3 = (const float*)d_in[17];
  float* recon = (float*)d_out;

  // Workspace layout (floats); bufA reused for h1 then d2, bufB for h2 then d1.
  float* ws   = (float*)d_ws;
  float* bufA = ws;                      // 32*64*128*128 = 33,554,432
  float* bufB = bufA + 33554432;         // 32*128*64*64 = 16,777,216
  float* ze   = bufB + 16777216;         // 32*64*32*32  =  2,097,152
  float* zq   = ze + 2097152;            //                2,097,152
  bf16_t* w1 = (bf16_t*)(zq + 2097152);  // [64][64]      (K=48 padded to 64)
  bf16_t* w2 = w1 + 64 * 64;             // [128][1024]
  bf16_t* w3 = w2 + 128 * 1024;          // [64][2048]
  bf16_t* t1 = w3 + 64 * 2048;           // [4][128][256]
  bf16_t* t2 = t1 + 4 * 128 * 256;       // [4][64][512]
  bf16_t* t3 = t2 + 4 * 64 * 512;        // [4][16][256]  (COUT=3 padded)
  float* stats1 = (float*)(t3 + 4 * 16 * 256);   // 2*64
  float* stats2 = stats1 + 128;                  // 2*128
  float* sse    = stats2 + 256;                  // 1

  // Per-launch init of accumulators (graph-replay safe).
  zero_ws<<<2, 256, 0, stream>>>(stats1, 128 + 256 + 1);

  // Weight conversion to bf16 GEMM layouts.
  prep_conv_w<3, 64><<<(64 * 64 + 255) / 256, 256, 0, stream>>>(ew1, w1);
  prep_conv_w<64, 128><<<(128 * 1024 + 255) / 256, 256, 0, stream>>>(ew2, w2);
  prep_conv_w<128, 64><<<(64 * 2048 + 255) / 256, 256, 0, stream>>>(ew3, w3);
  prep_convt_w<64, 128, 128><<<(4 * 128 * 256 + 255) / 256, 256, 0, stream>>>(dw1, t1);
  prep_convt_w<128, 64, 64><<<(4 * 64 * 512 + 255) / 256, 256, 0, stream>>>(dw2, t2);
  prep_convt_w<64, 3, 16><<<(4 * 16 * 256 + 255) / 256, 256, 0, stream>>>(dw3, t3);

  // ---- encoder ----
  conv_s2_wmma<3, 64, 256, 256, 4, 1><<<dim3(8, 32 * 128, 1), 128, 0, stream>>>(x, w1, eb1, bufA);
  bn_stats<64, 16384><<<dim3(64, 32), 256, 0, stream>>>(bufA, stats1);
  bn_apply_silu<64, 16384, 32><<<(33554432 + 255) / 256, 256, 0, stream>>>(bufA, stats1, bg1, bb1);

  conv_s2_wmma<64, 128, 128, 128, 4, 2><<<dim3(4, 32 * 64, 1), 128, 0, stream>>>(bufA, w2, eb2, bufB);
  bn_stats<128, 4096><<<dim3(128, 32), 256, 0, stream>>>(bufB, stats2);
  bn_apply_silu<128, 4096, 32><<<(16777216 + 255) / 256, 256, 0, stream>>>(bufB, stats2, bg2, bb2);

  conv_s2_wmma<128, 64, 64, 64, 2, 2><<<dim3(2, 32 * 32, 1), 64, 0, stream>>>(bufB, w3, eb3, ze);

  // ---- vector quantizer ----
  vq_argmin<<<32768, 32, 0, stream>>>(ze, cb, zq, sse);

  // ---- decoder ----
  convt_s2_wmma<64, 128, 128, 32, 32, 0, 4, 2><<<dim3(2, 32 * 32, 4), 128, 0, stream>>>(zq, t1, db1, bufB);
  convt_s2_wmma<128, 64, 64, 64, 64, 0, 2, 2><<<dim3(4, 32 * 64, 4), 64, 0, stream>>>(bufB, t2, db2, bufA);
  convt_s2_wmma<64, 3, 16, 128, 128, 1, 1, 1><<<dim3(8, 32 * 128, 4), 32, 0, stream>>>(bufA, t3, db3, recon);

  // ---- loss ----
  finish_loss<<<1, 32, 0, stream>>>(sse, recon + 6291456);
}
